// LinearCrossAttention_60816736911886
// MI455X (gfx1250) — compile-verified, matrix operand order checked by reading, not run on hardware
//
#include <hip/hip_runtime.h>

typedef __attribute__((ext_vector_type(16))) _Float16 v16h;
typedef __attribute__((ext_vector_type(8)))  _Float16 v8h;
typedef __attribute__((ext_vector_type(8)))  float    v8f;
typedef unsigned int u32x4 __attribute__((ext_vector_type(4)));
typedef int          i32x4 __attribute__((ext_vector_type(4)));
typedef int          i32x8 __attribute__((ext_vector_type(8)));

#define N_TOK  16384
#define DIM    72
#define DIMP   96            // DIM padded to 3*32 WMMA K-steps; col 72 of K = 1.0 (row-sum trick)
#define POSF   16
#define CS     512
#define NB     (N_TOK / CS)  // 32 k-blocks
#define JTILE  128           // k-rows staged in LDS per barrier; softmax update once per tile

#if defined(__has_builtin)
#if __has_builtin(__builtin_amdgcn_tensor_load_to_lds) && __has_builtin(__builtin_amdgcn_s_wait_tensorcnt)
#define USE_TDM 1
#endif
#endif

// LDS layout inside a single shared block (single allocation => base offset 0)
#define SM_KKR 0                         // [128][96] f16 row-major k-tile   (24576 B)
#define SM_KKT 24576                     // [96][128] f16 transposed k-tile  (24576 B)
#define SM_PST 49152                     // 8 waves x [16][32] f16 P staging ( 8192 B)
#define SM_TOTAL 57344

// ---------------------------------------------------------------------------
// Phase 1: fused positional-encoding MLP + projection -> f16, padded to 96.
// Column 72 gets `pad72` (1.0 for K so the PV WMMA also produces row sums,
// 0.0 for Q so QK^T is exact); 73..95 are 0. Optionally also writes the
// transposed matrix outT[96][N_TOK] for contiguous transposed staging.
// ---------------------------------------------------------------------------
__global__ void proj_pos_kernel(const float* __restrict__ x,
                                const float* __restrict__ coord,
                                const float* __restrict__ W,
                                const float* __restrict__ bias,
                                const float* __restrict__ pe_w1,
                                const float* __restrict__ pe_b1,
                                const float* __restrict__ pe_w2,
                                const float* __restrict__ pe_b2,
                                _Float16* __restrict__ outh,
                                _Float16* __restrict__ outT,
                                float pad72) {
    const int row = blockIdx.x;
    const int t = threadIdx.x;
    __shared__ float xrow[DIM];
    __shared__ float h[POSF];
    if (t < DIM) xrow[t] = x[(size_t)row * DIM + t];
    if (t < POSF) {
        float c0 = coord[row * 2 + 0], c1 = coord[row * 2 + 1];
        float z = c0 * pe_w1[t * 2 + 0] + c1 * pe_w1[t * 2 + 1] + pe_b1[t];
        h[t] = 0.5f * z * (1.0f + erff(z * 0.70710678118654752f));  // exact GELU
    }
    __syncthreads();
    _Float16 hv;
    if (t < DIM) {
        float acc = bias[t] + pe_b2[t];
        #pragma unroll
        for (int p = 0; p < POSF; ++p) acc += h[p] * pe_w2[t * POSF + p];
        for (int c = 0; c < DIM; ++c) acc += xrow[c] * W[t * DIM + c];
        hv = (_Float16)acc;
    } else {
        hv = (t == DIM) ? (_Float16)pad72 : (_Float16)0.0f;
    }
    outh[(size_t)row * DIMP + t] = hv;
    if (outT) outT[(size_t)t * N_TOK + row] = hv;
}

#ifdef USE_TDM
// Issue a 2-D TDM tile load: global (elem=2B) -> LDS, row-major packing.
// D# per CDNA5 ISA (count=1, type=2 "image").
// 6-arg builtin form (clang-23 / therock-10.0 headers).
__device__ __forceinline__ void tdm_load_2d(unsigned lds_byte_off,
                                            unsigned long long gaddr,
                                            unsigned tensor_d0, unsigned tensor_d1,
                                            unsigned tile_d0, unsigned tile_d1,
                                            unsigned long long stride0_elems) {
    u32x4 g0;
    g0.x = 1u;                                                    // count=1 valid user D#
    g0.y = lds_byte_off;                                          // lds_addr
    g0.z = (unsigned)(gaddr & 0xFFFFFFFFull);                     // global_addr[31:0]
    g0.w = (unsigned)((gaddr >> 32) & 0x01FFFFFFull) | 0x80000000u; // addr[56:32] | type=2
    i32x8 g1;
    g1[0] = (int)(1u << 16);                                      // data_size=1 (2 bytes)
    g1[1] = (int)((tensor_d0 & 0xFFFFu) << 16);                   // tensor_dim0[15:0]
    g1[2] = (int)(((tensor_d0 >> 16) & 0xFFFFu) | ((tensor_d1 & 0xFFFFu) << 16));
    g1[3] = (int)(((tensor_d1 >> 16) & 0xFFFFu) | ((tile_d0 & 0xFFFFu) << 16));
    g1[4] = (int)(tile_d1 & 0xFFFFu);                             // tile_dim2=0
    g1[5] = (int)(stride0_elems & 0xFFFFFFFFull);                 // dim0 stride
    g1[6] = (int)((stride0_elems >> 32) & 0xFFFFull);             // stride hi | stride1=0
    g1[7] = 0;
    i32x4 z4 = {0, 0, 0, 0};
    i32x8 z8 = {0, 0, 0, 0, 0, 0, 0, 0};
    __builtin_amdgcn_tensor_load_to_lds(g0, g1, z4, z4, z8, 0);
}
#endif

// ---------------------------------------------------------------------------
// Phase 2: streaming blockwise attention, per-block online softmax with ONE
// max/rescale update per 128-column LDS tile. PV WMMA column 72 (==1.0)
// accumulates the softmax denominators as a byproduct.
// ---------------------------------------------------------------------------
__global__ __launch_bounds__(256) void attn_kernel(const _Float16* __restrict__ qh,
                                                   const _Float16* __restrict__ kh,
                                                   const _Float16* __restrict__ khT,
                                                   float* __restrict__ out) {
    __shared__ alignas(32) unsigned char smem[SM_TOTAL];
    _Float16* kkR = (_Float16*)&smem[SM_KKR];           // [JTILE][DIMP]
    _Float16* kkT = (_Float16*)&smem[SM_KKT];           // [DIMP][JTILE]
    _Float16* pst = (_Float16*)&smem[SM_PST];           // [8][16*32]

    const int tid  = threadIdx.x;
    const int lane = tid & 31;            // wave32
    const int wave = tid >> 5;
    const int lhi  = lane >> 4;
    const int llo  = lane & 15;
    const int row0 = blockIdx.x * 128 + wave * 16;

    const v8f VZERO = {0.f, 0.f, 0.f, 0.f, 0.f, 0.f, 0.f, 0.f};

    // Persistent A-fragments of this wave's 16 q-rows (3 K-steps of 32)
    v16h aq[3];
    #pragma unroll
    for (int ks = 0; ks < 3; ++ks)
        aq[ks] = *(const v16h*)(qh + (size_t)(row0 + llo) * DIMP + ks * 32 + lhi * 16);

    v8f oacc[5];
    #pragma unroll
    for (int t = 0; t < 5; ++t) oacc[t] = VZERO;

    for (int m = 0; m < NB; ++m) {
        float mrow[8];
        v8f acc[5];
        #pragma unroll
        for (int r = 0; r < 8; ++r) mrow[r] = -__builtin_inff();
        #pragma unroll
        for (int t = 0; t < 5; ++t) acc[t] = VZERO;

        for (int q4 = 0; q4 < CS / JTILE; ++q4) {
            const int jbase = m * CS + q4 * JTILE;
            __syncthreads();                      // previous tile fully consumed
#ifdef USE_TDM
            if (tid < 32) {                       // wave 0 drives the Tensor Data Mover
                tdm_load_2d(SM_KKR,
                            (unsigned long long)(kh + (size_t)jbase * DIMP),
                            DIMP, N_TOK, DIMP, JTILE, DIMP);
                tdm_load_2d(SM_KKT,
                            (unsigned long long)(khT + (size_t)jbase),
                            N_TOK, DIMP, JTILE, DIMP, N_TOK);
                __builtin_amdgcn_s_wait_tensorcnt(0);
            }
#else
            for (int id = tid; id < JTILE * (DIMP / 8); id += 256) {   // row-major tile
                int j = id / (DIMP / 8), cg = id % (DIMP / 8);
                *(v8h*)&kkR[j * DIMP + cg * 8] =
                    *(const v8h*)(kh + (size_t)(jbase + j) * DIMP + cg * 8);
            }
            for (int id = tid; id < DIMP * (JTILE / 8); id += 256) {   // transposed tile
                int c = id / (JTILE / 8), cg = id % (JTILE / 8);
                *(v8h*)&kkT[c * JTILE + cg * 8] =
                    *(const v8h*)(khT + (size_t)c * N_TOK + jbase + cg * 8);
            }
#endif
            __syncthreads();

            // ---- S = Q · K^T for ALL 128 columns of this tile (8 sub-tiles) ----
            v8f s[8];
            #pragma unroll
            for (int idx = 0; idx < 8; ++idx) {
                const int js = idx >> 1, ct = idx & 1;
                s[idx] = VZERO;
                #pragma unroll
                for (int ks = 0; ks < 3; ++ks) {
                    v16h b = *(const v16h*)&kkR[(js * 32 + ct * 16 + llo) * DIMP +
                                                ks * 32 + lhi * 16];
                    s[idx] = __builtin_amdgcn_wmma_f32_16x16x32_f16(
                        false, aq[ks], false, b, (short)0, s[idx], false, false);
                }
            }

            // ---- ONE online-max update over all 128 columns ----
            float mnew[8], alpha[8];
            #pragma unroll
            for (int r = 0; r < 8; ++r) {
                float v = s[0][r];
                #pragma unroll
                for (int idx = 1; idx < 8; ++idx) v = fmaxf(v, s[idx][r]);
                v = fmaxf(v, __shfl_xor(v, 1, 32));
                v = fmaxf(v, __shfl_xor(v, 2, 32));
                v = fmaxf(v, __shfl_xor(v, 4, 32));
                v = fmaxf(v, __shfl_xor(v, 8, 32));
                mnew[r]  = fmaxf(mrow[r], v);
                alpha[r] = __expf(mrow[r] - mnew[r]);   // first tile: exp(-inf)=0
                mrow[r]  = mnew[r];
            }
            #pragma unroll
            for (int t = 0; t < 5; ++t)
                #pragma unroll
                for (int r = 0; r < 8; ++r)
                    acc[t][r] *= alpha[r];

            // exponentiate in place: s becomes P
            #pragma unroll
            for (int idx = 0; idx < 8; ++idx)
                #pragma unroll
                for (int r = 0; r < 8; ++r)
                    s[idx][r] = __expf(s[idx][r] - mnew[r]);

            // ---- P · K in 4 chunks of 32 columns; col 72 accumulates row sums ----
            #pragma unroll
            for (int js = 0; js < JTILE / 32; ++js) {
                #pragma unroll
                for (int ct = 0; ct < 2; ++ct)
                    #pragma unroll
                    for (int r = 0; r < 8; ++r)
                        pst[wave * 512 + (r + 8 * lhi) * 32 + ct * 16 + llo] =
                            (_Float16)s[js * 2 + ct][r];
                v16h pa = *(const v16h*)&pst[wave * 512 + llo * 32 + lhi * 16];
                #pragma unroll
                for (int nt = 0; nt < 5; ++nt) {
                    v16h b = *(const v16h*)&kkT[(nt * 16 + llo) * JTILE +
                                                js * 32 + lhi * 16];
                    acc[nt] = __builtin_amdgcn_wmma_f32_16x16x32_f16(
                        false, pa, false, b, (short)0, acc[nt], false, false);
                }
            }
        }

        // normalize by the WMMA-produced row sums (col 72 -> lane llo==8 of acc[4])
        #pragma unroll
        for (int r = 0; r < 8; ++r) {
            float l  = __shfl(acc[4][r], (lane & 16) + 8, 32);
            float rl = 1.0f / l;
            #pragma unroll
            for (int t = 0; t < 5; ++t) oacc[t][r] += acc[t][r] * rl;
        }
    }

    // write result, dropping padded columns 72..79
    #pragma unroll
    for (int nt = 0; nt < 5; ++nt) {
        int col = nt * 16 + llo;
        if (col < DIM) {
            #pragma unroll
            for (int r = 0; r < 8; ++r) {
                int row = row0 + r + 8 * lhi;
                out[(size_t)row * DIM + col] = oacc[nt][r];
            }
        }
    }
}

extern "C" void kernel_launch(void* const* d_in, const int* in_sizes, int n_in,
                              void* d_out, int out_size, void* d_ws, size_t ws_size,
                              hipStream_t stream) {
    const float* q        = (const float*)d_in[0];
    const float* kv       = (const float*)d_in[1];
    const float* coord_q  = (const float*)d_in[2];
    const float* coord_kv = (const float*)d_in[3];
    const float* pe_w1    = (const float*)d_in[4];
    const float* pe_b1    = (const float*)d_in[5];
    const float* pe_w2    = (const float*)d_in[6];
    const float* pe_b2    = (const float*)d_in[7];
    const float* wq       = (const float*)d_in[8];
    const float* bq       = (const float*)d_in[9];
    const float* wk       = (const float*)d_in[10];
    const float* bk       = (const float*)d_in[11];
    // d_in[12]/d_in[13] (wv, bv): reference computes v but never uses it.

    _Float16* qh  = (_Float16*)d_ws;                       // [16384 x 96]
    _Float16* kh  = qh + (size_t)N_TOK * DIMP;             // [16384 x 96]
    _Float16* khT = kh + (size_t)N_TOK * DIMP;             // [96 x 16384]
    float* out = (float*)d_out;

    proj_pos_kernel<<<N_TOK, DIMP, 0, stream>>>(q,  coord_q,  wq, bq,
                                                pe_w1, pe_b1, pe_w2, pe_b2,
                                                qh, nullptr, 0.0f);
    proj_pos_kernel<<<N_TOK, DIMP, 0, stream>>>(kv, coord_kv, wk, bk,
                                                pe_w1, pe_b1, pe_w2, pe_b2,
                                                kh, khT, 1.0f);
    attn_kernel<<<N_TOK / 128, 256, 0, stream>>>(qh, kh, khT, out);
}
